// MultiABSA_13821204758665
// MI455X (gfx1250) — matrix operand order repaired and checked by reading, not compiled, not run on hardware
//
#include <hip/hip_runtime.h>
#include <math.h>

// ---------------- problem constants ----------------
constexpr int B_   = 64;
constexpr int T_   = 512;
constexpr int E_   = 300;
constexpr int NC_  = 5;
constexpr int ROWS = B_ * T_;      // 32768
constexpr int GS   = 480;          // padded 3H (=450) stride

// ---------------- WMMA types / helpers ----------------
typedef __attribute__((ext_vector_type(16))) __bf16 v16bf;
typedef __attribute__((ext_vector_type(4)))  __bf16 v4bf;
typedef __attribute__((ext_vector_type(2)))  __bf16 v2bf;
typedef __attribute__((ext_vector_type(8)))  float  v8f;
typedef unsigned int v4u __attribute__((ext_vector_type(4)));
typedef int          v8i __attribute__((ext_vector_type(8)));

__device__ static inline v8f wmma_bf16(v16bf a, v16bf b, v8f c) {
  return __builtin_amdgcn_wmma_f32_16x16x32_bf16(false, a, false, b, (short)0, c, false, false);
}

// Inverse fragment swizzles (16-bit A 16x32 / B 32x16 in contiguous k order, half = k/16, elem = k%16; 32-bit C 16x16 layouts).
// Groups of 4 consecutive K (4-aligned) stay in one lane at consecutive elements.
__device__ static inline int a_lane(int row, int kk) { return (row & 15) + (kk & 16); }
__device__ static inline int a_elem(int kk)          { return kk & 15; }
__device__ static inline int b_lane(int col, int kk) { return (col & 15) + (kk & 16); }

__device__ static inline v4bf pack4(float4 v, bool ok) {
  v4bf p;
  p[0] = (__bf16)(ok ? v.x : 0.f);
  p[1] = (__bf16)(ok ? v.y : 0.f);
  p[2] = (__bf16)(ok ? v.z : 0.f);
  p[3] = (__bf16)(ok ? v.w : 0.f);
  return p;
}

__device__ static inline float sigmoidf_(float x) { return 1.f / (1.f + expf(-x)); }

// ---------------- K1: stable descending argsort of lengths (B=64) ----------------
__global__ __launch_bounds__(64) void k_sort(const int* __restrict__ len,
                                             int* __restrict__ perm, int* __restrict__ Ls) {
  int i = threadIdx.x;
  int Li = len[i];
  int rank = 0;
  for (int j = 0; j < 64; ++j) {
    int Lj = len[j];
    rank += (Lj > Li) || (Lj == Li && j < i);
  }
  perm[rank] = i;
  Ls[rank]   = Li;
}

// ---------------- K2: input projection GEMM  xp[c,dir] = x_sorted @ W_ih^T + b_ih ----------------
__global__ __launch_bounds__(256) void k_inproj(
    const float* __restrict__ docs, const int* __restrict__ perm,
    const float* __restrict__ wihf, const float* __restrict__ wihb,
    const float* __restrict__ bihf, const float* __restrict__ bihb,
    float* __restrict__ xpf, float* __restrict__ xpb)
{
  __shared__ __attribute__((aligned(32))) __bf16 Afrag[4 * 512];
  __shared__ __attribute__((aligned(32))) __bf16 Bfrag[2 * 512];
  __shared__ int prm[64];

  const int t   = blockIdx.x;
  const int n0  = blockIdx.y * 32;
  const int cd  = blockIdx.z;
  const int c   = cd >> 1, dir = cd & 1;
  const float* wih = (dir ? wihb : wihf) + (size_t)c * 450 * 300;
  const float* bih = (dir ? bihb : bihf) + c * 450;
  float*       xp  = (dir ? xpb  : xpf)  + (size_t)c * ROWS * GS;

  const int tid  = threadIdx.x;
  const int lane = tid & 31;
  const int wave = tid >> 5;
  const int msub = wave & 3, nsub = wave >> 2;

  if (tid < 64) prm[tid] = perm[tid];
  __syncthreads();

  // per-thread staging assignments (constant across k-steps)
  // A: 512 groups of 4k; thread handles g = tid and tid+256
  const int rowA0 = tid >> 3,        rowA1 = rowA0 + 32;
  const int kgA   = (tid & 7) * 4;   // within-tile k of first element
  const float* aSrc0 = docs + ((size_t)prm[rowA0] * T_ + t) * E_;
  const float* aSrc1 = docs + ((size_t)prm[rowA1] * T_ + t) * E_;
  const int aDst0 = (rowA0 >> 4) * 512 + a_lane(rowA0, kgA) * 16 + a_elem(kgA);
  const int aDst1 = (rowA1 >> 4) * 512 + a_lane(rowA1, kgA) * 16 + a_elem(kgA);
  // B: 256 groups; thread handles one: col j = tid>>3, k group = (tid&7)*4
  const int jB   = tid >> 3;
  const int colB = n0 + jB;
  const bool colOK = (colB < 450);
  const float* bSrc = wih + (size_t)(colOK ? colB : 449) * 300;
  const int bDst = (jB >> 4) * 512 + b_lane(jB, kgA) * 16 + (kgA & 15);

  v8f acc = {};
  for (int ks = 0; ks < 10; ++ks) {
    int k = ks * 32 + kgA;
    int kc = (k <= 296) ? k : 296;     // 300 % 4 == 0 -> group validity uniform
    bool kOK = (k < 300);
    float4 va0 = *(const float4*)(aSrc0 + kc);
    float4 va1 = *(const float4*)(aSrc1 + kc);
    float4 vb  = *(const float4*)(bSrc + kc);
    if (ks + 1 < 10) __builtin_prefetch(aSrc0 + kc + 32, 0, 0);
    *(v4bf*)&Afrag[aDst0] = pack4(va0, kOK);
    *(v4bf*)&Afrag[aDst1] = pack4(va1, kOK);
    *(v4bf*)&Bfrag[bDst]  = pack4(vb, kOK && colOK);
    __syncthreads();
    v16bf a = *(const v16bf*)&Afrag[msub * 512 + lane * 16];
    v16bf b = *(const v16bf*)&Bfrag[nsub * 512 + lane * 16];
    acc = wmma_bf16(a, b, acc);
    __syncthreads();
  }
  int colg = n0 + nsub * 16 + (lane & 15);
  float bias = (colg < 450) ? bih[colg] : 0.f;
  int rbase = t * 64 + msub * 16 + ((lane >> 4) * 8);
#pragma unroll
  for (int r = 0; r < 8; ++r)
    xp[(size_t)(rbase + r) * GS + colg] = acc[r] + bias;
}

// ---------------- K3: GRU scan, one workgroup per (c,dir); weights + state resident in 320KB LDS ----
// LDS map (dynamic, 326144 B):
//   wBf [5 ks][29 n][512] bf16 (B-frag order)   @      0 .. 148480
//   hS  [64][150] f32 (carried state)           @ 148480 .. 186880
//   hA  [4 m][5 ks][512] bf16 (A-frag order)    @ 186880 .. 207360
//   ghf [4 m][29 n][256] f32 (C-frag order)     @ 207360 .. 326144
__global__ __launch_bounds__(1024) void k_scan(
    const float* __restrict__ whhf, const float* __restrict__ whhb,
    const float* __restrict__ bhhf, const float* __restrict__ bhhb,
    const float* __restrict__ xpf,  const float* __restrict__ xpb,
    const int*   __restrict__ Ls,   float* __restrict__ hall)
{
  extern __shared__ char smem[];
  __bf16* wBf = (__bf16*)smem;
  float*  hS  = (float*)(smem + 148480);
  __bf16* hA  = (__bf16*)(smem + 186880);
  float*  ghf = (float*)(smem + 207360);
  __shared__ int Lsm[64];

  const int blk = blockIdx.x;
  const int c = blk >> 1, dir = blk & 1;
  const float* whh = (dir ? whhb : whhf) + (size_t)c * 450 * 150;
  const float* bhh = (dir ? bhhb : bhhf) + c * 450;
  const float* xp  = (dir ? xpb  : xpf)  + (size_t)c * ROWS * GS;

  const int tid = threadIdx.x;
  const int lane = tid & 31;
  const int wave = tid >> 5;

  // resident W_hh in B-fragment order, float2-paired loads (150 row stride -> 8B aligned)
  for (int g = tid; g < 80 * 464; g += 1024) {
    int k2 = g / 464, n = g - k2 * 464;
    int k = k2 * 2;
    int kc = (k < 150) ? k : 148;
    int nc = (n < 450) ? n : 449;
    float2 v = *(const float2*)&whh[(size_t)nc * 150 + kc];
    bool ok = (k < 150) && (n < 450);      // 150 even -> pair validity uniform
    v2bf p;
    p[0] = (__bf16)(ok ? v.x : 0.f);
    p[1] = (__bf16)(ok ? v.y : 0.f);
    *(v2bf*)&wBf[((k >> 5) * 29 + (n >> 4)) * 512 + b_lane(n, k & 31) * 16 + (k & 15)] = p;
  }
  for (int idx = tid; idx < 4 * 5 * 512; idx += 1024) hA[idx] = (__bf16)0.f;
  for (int idx = tid; idx < 64 * 150; idx += 1024) hS[idx] = 0.f;
  if (tid < 64) Lsm[tid] = Ls[tid];
  __syncthreads();

  for (int s = 0; s < T_; ++s) {
    // GEMM: gh = h @ W_hh^T  (M=64,N=464,K=160); 4x29 tiles over 32 waves
    for (int tau = wave; tau < 116; tau += 32) {
      int m = tau & 3, n = tau >> 2;
      v8f acc = {};
#pragma unroll
      for (int ks = 0; ks < 5; ++ks) {
        v16bf a = *(const v16bf*)&hA[(m * 5 + ks) * 512 + lane * 16];
        v16bf b = *(const v16bf*)&wBf[(ks * 29 + n) * 512 + lane * 16];
        acc = wmma_bf16(a, b, acc);
      }
      *(v8f*)&ghf[(m * 29 + n) * 256 + lane * 8] = acc;
    }
    __syncthreads();
    // gate phase: 64*150 elements
    for (int idx = tid; idx < 64 * 150; idx += 1024) {
      int b = idx / 150, j = idx - b * 150;
      int L = Lsm[b];
      int tst = dir ? ((s < L) ? (L - 1 - s) : s) : s;
      const float* xr = &xp[(size_t)(tst * 64 + b) * GS];
      int m = b >> 4, lhalf = ((b >> 3) & 1) * 16, rr = b & 7;
      auto ghrd = [&](int col) {
        return ghf[(m * 29 + (col >> 4)) * 256 + ((col & 15) + lhalf) * 8 + rr];
      };
      float rg = sigmoidf_(xr[j]       + ghrd(j)       + bhh[j]);
      float zg = sigmoidf_(xr[150 + j] + ghrd(150 + j) + bhh[150 + j]);
      float ng = tanhf   (xr[300 + j] + rg * (ghrd(300 + j) + bhh[300 + j]));
      float hnew = (1.f - zg) * ng + zg * hS[b * 150 + j];
      hS[b * 150 + j] = hnew;
      int kk = j & 31;
      hA[(m * 5 + (j >> 5)) * 512 + a_lane(b, kk) * 16 + a_elem(kk)] = (__bf16)hnew;
      hall[(((size_t)c * 64 + b) * 512 + tst) * 300 + dir * 150 + j] = hnew;
    }
    __syncthreads();
  }
}

// ---------------- K4: TDM-staged LayerNorm + sentence attention (WMMA) + task logits ----------------
// LDS map (dynamic, 232448 B):
//   hnf  [4 m][10 ks][512] bf16 (A-frag order) @      0 ..  40960
//   Wlf  [10 ks][8 n][512] bf16 (B-frag order) @  40960 .. 122880
//   uL   [64][128] f32                          @ 122880 .. 155648
//   hraw [64][300] f32 (TDM tile)               @ 155648 .. 232448
__global__ __launch_bounds__(256) void k_lnatt(
    const float* __restrict__ hall, const float* __restrict__ gamma,
    const float* __restrict__ beta, const float* __restrict__ wsent,
    const float* __restrict__ bsent, const float* __restrict__ taskw,
    float* __restrict__ ebuf)
{
  extern __shared__ char smem[];
  __bf16* hnf  = (__bf16*)smem;
  __bf16* Wlf  = (__bf16*)(smem + 40960);
  float*  uL   = (float*)(smem + 122880);

  const int rb = blockIdx.x;               // 0..511 (64-row blocks)
  const int c  = blockIdx.y;               // channel
  const int row0 = rb * 64;
  const int tid = threadIdx.x, lane = tid & 31, wave = tid >> 5;

  // --- Tensor Data Mover: DMA the 64x300 f32 h_all tile into LDS (wave 0 issues) ---
  if (wave == 0) {
    unsigned long long ga =
        (unsigned long long)(const void*)&hall[((size_t)c * ROWS + row0) * 300];
    v4u g0;
    g0.x = 1u;                                   // count=1 valid user descriptor
    g0.y = 155648u;                              // lds_addr (dynamic LDS base == 0, no static LDS)
    g0.z = (unsigned)ga;                         // global_addr[31:0]
    g0.w = (unsigned)(ga >> 32) | (2u << 30);    // global_addr[56:32] | type=2 (image)
    v8i g1;
    g1[0] = 0x20000;                             // data_size=2 -> 4-byte elements
    g1[1] = (int)(300u << 16);                   // tensor_dim0 = 300
    g1[2] = (int)(64u << 16);                    // tensor_dim1 = 64
    g1[3] = (int)(300u << 16);                   // tile_dim0 = 300
    g1[4] = 64;                                  // tile_dim1 = 64, tile_dim2 = 0
    g1[5] = 300;                                 // tensor_dim0_stride = 300
    g1[6] = 0;
    g1[7] = 0;
    asm volatile("tensor_load_to_lds %0, %1" :: "s"(g0), "s"(g1) : "memory");
  }
  // stage W_sent in B-fragment order while the DMA runs (float4 groups of 4k)
  for (int g = tid; g < 80 * 128; g += 256) {
    int kg = g >> 7, n = g & 127;
    int k = kg * 4;
    int kc = (k <= 296) ? k : 296;
    float4 v = *(const float4*)&wsent[((size_t)c * 128 + n) * 300 + kc];
    *(v4bf*)&Wlf[((k >> 5) * 8 + (n >> 4)) * 512 + b_lane(n, k & 31) * 16 + (k & 15)] =
        pack4(v, k < 300);
  }
  __builtin_amdgcn_s_wait_tensorcnt(0);
  __syncthreads();

  // LayerNorm one row per thread (float4 LDS reads), write hn in A-fragment order
  const float* hraw = (const float*)(smem + 155648);
  if (tid < 64) {
    const float4* hp4 = (const float4*)&hraw[tid * 300];
    float s = 0.f, s2 = 0.f;
    for (int k4 = 0; k4 < 75; ++k4) {
      float4 v = hp4[k4];
      s  += v.x + v.y + v.z + v.w;
      s2 += v.x * v.x + v.y * v.y + v.z * v.z + v.w * v.w;
    }
    float mu  = s / 300.f;
    float var = s2 / 300.f - mu * mu;
    float rstd = rsqrtf(var + 1e-5f);
    int m = tid >> 4;
    const float4* gm4 = (const float4*)&gamma[c * 300];
    const float4* bt4 = (const float4*)&beta[c * 300];
    for (int kg = 0; kg < 80; ++kg) {
      int k = kg * 4, kk = k & 31;
      int dst = (m * 10 + (k >> 5)) * 512 + a_lane(tid, kk) * 16 + a_elem(kk);
      v4bf p;
      if (kg < 75) {
        float4 h = hp4[kg], gm = gm4[kg], bt = bt4[kg];
        p[0] = (__bf16)((h.x - mu) * rstd * gm.x + bt.x);
        p[1] = (__bf16)((h.y - mu) * rstd * gm.y + bt.y);
        p[2] = (__bf16)((h.z - mu) * rstd * gm.z + bt.z);
        p[3] = (__bf16)((h.w - mu) * rstd * gm.w + bt.w);
      } else {
        p[0] = p[1] = p[2] = p[3] = (__bf16)0.f;
      }
      *(v4bf*)&hnf[dst] = p;
    }
  }
  __syncthreads();

  // u = tanh(hn @ W^T + b): M=64,N=128,K=320; 4x8 tiles over 8 waves
#pragma unroll
  for (int tt = 0; tt < 4; ++tt) {
    int tau = wave + tt * 8;
    int m = tau & 3, n = tau >> 2;
    v8f acc = {};
#pragma unroll
    for (int ks = 0; ks < 10; ++ks) {
      v16bf a = *(const v16bf*)&hnf[(m * 10 + ks) * 512 + lane * 16];
      v16bf b = *(const v16bf*)&Wlf[(ks * 8 + n) * 512 + lane * 16];
      acc = wmma_bf16(a, b, acc);
    }
    int col = n * 16 + (lane & 15);
    float bias = bsent[c * 128 + col];
    int rbase = m * 16 + ((lane >> 4) * 8);
#pragma unroll
    for (int r = 0; r < 8; ++r)
      uL[(rbase + r) * 128 + col] = tanhf(acc[r] + bias);
  }
  __syncthreads();

  // task logits: e[k,row] = u[row,:] . task_cv_w[k,:]  (float4 dots)
  const int CVS[5] = {0, 9, 12, 18, 24};
  const int CVC[5] = {9, 3, 6, 6, 6};
  int start = CVS[c], cnt = CVC[c];
  for (int idx = tid; idx < cnt * 64; idx += 256) {
    int kk = idx >> 6, rl = idx & 63;
    const float4* u4  = (const float4*)&uL[rl * 128];
    const float4* tw4 = (const float4*)&taskw[(size_t)(start + kk) * 128];
    float acc = 0.f;
    for (int a4 = 0; a4 < 32; ++a4) {
      float4 u = u4[a4], w = tw4[a4];
      acc += u.x * w.x + u.y * w.y + u.z * w.z + u.w * w.w;
    }
    ebuf[(size_t)(start + kk) * ROWS + row0 + rl] = acc;
  }
}

// ---------------- K5a: global max over (b,t) per task k (masked) ----------------
__global__ __launch_bounds__(256) void k_gmax(const float* __restrict__ e,
                                              const int* __restrict__ Ls,
                                              float* __restrict__ gmax) {
  __shared__ float red[256];
  int k = blockIdx.x, tid = threadIdx.x;
  float m = -1e30f;
  for (int r = tid; r < ROWS; r += 256) {
    int b = r >> 9, t = r & 511;
    float v = e[(size_t)k * ROWS + r];
    m = (t < Ls[b]) ? fmaxf(m, v) : m;
  }
  red[tid] = m; __syncthreads();
  for (int s = 128; s > 0; s >>= 1) {
    if (tid < s) red[tid] = fmaxf(red[tid], red[tid + s]);
    __syncthreads();
  }
  if (tid == 0) gmax[k] = red[0];
}

// ---------------- K5b: masked softmax over t per (k,b) ----------------
__global__ __launch_bounds__(256) void k_att(const float* __restrict__ e,
                                             const int* __restrict__ Ls,
                                             const float* __restrict__ gmax,
                                             float* __restrict__ att) {
  __shared__ float red[256];
  int k = blockIdx.x, b = blockIdx.y, tid = threadIdx.x;
  int L = Ls[b];
  float g = gmax[k];
  const float* ep = &e[(size_t)k * ROWS + b * 512];
  float s = 0.f;
  for (int t = tid; t < 512; t += 256)
    s += (t < L) ? expf(ep[t] - g) : 0.f;
  red[tid] = s; __syncthreads();
  for (int st = 128; st > 0; st >>= 1) {
    if (tid < st) red[tid] += red[tid + st];
    __syncthreads();
  }
  float inv = 1.f / red[0];
  for (int t = tid; t < 512; t += 256)
    att[(size_t)k * ROWS + b * 512 + t] = (t < L) ? expf(ep[t] - g) * inv : 0.f;
}

// ---------------- K6: docv[k,b,:] = sum_t att[k,b,t] * h_all[4][b,t,:] ----------------
__global__ __launch_bounds__(320) void k_docv(const float* __restrict__ att,
                                              const float* __restrict__ hall,
                                              float* __restrict__ docv) {
  int k = blockIdx.x, b = blockIdx.y, d = threadIdx.x;
  if (d >= 300) return;
  const float4* ap4 = (const float4*)&att[(size_t)k * ROWS + b * 512];
  const float* hp = &hall[(((size_t)4 * 64 + b) * 512) * 300 + d];
  float acc = 0.f;
  for (int t4 = 0; t4 < 128; ++t4) {
    float4 a4 = ap4[t4];
    const float* h0 = hp + (size_t)t4 * 4 * 300;
    acc += a4.x * h0[0] + a4.y * h0[300] + a4.z * h0[600] + a4.w * h0[900];
  }
  docv[((size_t)k * 64 + b) * 300 + d] = acc;
}

// ---------------- K7: group attention + weighted combine -> output (9,64,300) ----------------
__global__ __launch_bounds__(64) void k_out(const float* __restrict__ docv,
                                            const float* __restrict__ gaw,
                                            const float* __restrict__ gab,
                                            const float* __restrict__ gcv,
                                            float* __restrict__ out) {
  __shared__ float red[64];
  __shared__ float en[8];
  const int TS[9]   = {0, 3, 6, 9, 12, 18, 21, 24, 27};
  const int TCNT[9] = {3, 3, 3, 3, 6, 3, 3, 3, 3};
  const int TCAT[9] = {0, 0, 0, 1, 2, 3, 3, 4, 4};
  int t9 = blockIdx.x, b = blockIdx.y, g = threadIdx.x;
  int start = TS[t9], cnt = TCNT[t9], cat = TCAT[t9];

  for (int n = 0; n < cnt; ++n) {
    const float4* dp4 = (const float4*)&docv[((size_t)(start + n) * 64 + b) * 300];
    const float4* wp4 = (const float4*)&gaw[((size_t)cat * 64 + g) * 300];
    float acc = gab[cat * 64 + g];
    for (int i4 = 0; i4 < 75; ++i4) {
      float4 d = dp4[i4], w = wp4[i4];
      acc += d.x * w.x + d.y * w.y + d.z * w.z + d.w * w.w;
    }
    acc *= gcv[t9 * 64 + g];
    red[g] = acc; __syncthreads();
    for (int st = 32; st > 0; st >>= 1) {
      if (g < st) red[g] += red[g + st];
      __syncthreads();
    }
    if (g == 0) en[n] = tanhf(red[0]);
    __syncthreads();
  }
  float w[6], S = 0.f;
  for (int n = 0; n < cnt; ++n) { w[n] = expf(en[n]); S += w[n]; }
  float invS = 1.f / S;
  for (int i = g; i < 300; i += 64) {
    float acc = 0.f;
    for (int n = 0; n < cnt; ++n)
      acc += docv[((size_t)(start + n) * 64 + b) * 300 + i] * w[n];
    out[((size_t)t9 * 64 + b) * 300 + i] = acc * invS;
  }
}

// ---------------- launch ----------------
extern "C" void kernel_launch(void* const* d_in, const int* in_sizes, int n_in,
                              void* d_out, int out_size, void* d_ws, size_t ws_size,
                              hipStream_t stream) {
  const float* docs  = (const float*)d_in[0];
  const int*   lens  = (const int*)d_in[1];
  const float* wihf  = (const float*)d_in[2];
  const float* whhf  = (const float*)d_in[3];
  const float* bihf  = (const float*)d_in[4];
  const float* bhhf  = (const float*)d_in[5];
  const float* wihb  = (const float*)d_in[6];
  const float* whhb  = (const float*)d_in[7];
  const float* bihb  = (const float*)d_in[8];
  const float* bhhb  = (const float*)d_in[9];
  const float* gamma = (const float*)d_in[10];
  const float* beta  = (const float*)d_in[11];
  const float* wsent = (const float*)d_in[12];
  const float* bsent = (const float*)d_in[13];
  const float* gaw   = (const float*)d_in[14];
  const float* gab   = (const float*)d_in[15];
  const float* taskw = (const float*)d_in[16];
  const float* gcv   = (const float*)d_in[17];
  float* out = (float*)d_out;

  char* ws = (char*)d_ws;
  int*   perm = (int*)ws;
  int*   Ls   = (int*)(ws + 256);
  float* gmax = (float*)(ws + 512);
  size_t off  = 4096;
  const size_t xpsz = (size_t)NC_ * ROWS * GS * sizeof(float);
  float* xpf  = (float*)(ws + off); off += xpsz;
  float* xpb  = (float*)(ws + off); off += xpsz;
  float* hall = (float*)(ws + off); off += (size_t)NC_ * ROWS * 300 * sizeof(float);
  float* ebuf = (float*)(ws + off); off += (size_t)30 * ROWS * sizeof(float);
  float* att  = (float*)(ws + off); off += (size_t)30 * ROWS * sizeof(float);
  float* docv = (float*)(ws + off); off += (size_t)30 * 64 * 300 * sizeof(float);
  (void)ws_size; (void)in_sizes; (void)n_in; (void)out_size;

  k_sort  <<<1, 64, 0, stream>>>(lens, perm, Ls);
  k_inproj<<<dim3(512, 15, 10), 256, 0, stream>>>(docs, perm, wihf, wihb, bihf, bihb, xpf, xpb);
  k_scan  <<<10, 1024, 326144, stream>>>(whhf, whhb, bhhf, bhhb, xpf, xpb, Ls, hall);
  k_lnatt <<<dim3(512, 5), 256, 232448, stream>>>(hall, gamma, beta, wsent, bsent, taskw, ebuf);
  k_gmax  <<<30, 256, 0, stream>>>(ebuf, Ls, gmax);
  k_att   <<<dim3(30, 64), 256, 0, stream>>>(ebuf, Ls, gmax, att);
  k_docv  <<<dim3(30, 64), 320, 0, stream>>>(att, hall, docv);
  k_out   <<<dim3(9, 64), 64, 0, stream>>>(docv, gaw, gab, gcv, out);
}
